// Multi_freq_71201967833633
// MI455X (gfx1250) — matrix operand order repaired and checked by reading, not compiled, not run on hardware
//
#include <hip/hip_runtime.h>

// Multi-frequency attention block for MI455X (gfx1250), wave32 + WMMA f16.
// All big GEMMs run through v_wmma_f32_16x16x32_f16 (f32 accum).
// A-operand fragments: 2x16B contiguous runs (k = 16*(lane/16) + s, same k order as the B fragments).
// B-operand fragments: fragment-major LDS swizzle -> ds_load_b128.
// Score buffer: column-major (stride 20 f32) so C-tile writeback is 2x ds_store_b128.
// Prefetches use max locality -> WGP-scope (fills near caches, ISA §10.5).
// Workspace requirement: ~806 MB.

#define Dm 512
#define Hh 8
#define Bb 4
#define Ss 48
#define Tt 300
#define TP 304   // T padded to multiple of 16

typedef __attribute__((ext_vector_type(16))) _Float16 v16h;
typedef __attribute__((ext_vector_type(8)))  _Float16 v8h;
typedef __attribute__((ext_vector_type(8)))  float    v8f;
typedef __attribute__((ext_vector_type(4)))  float    v4f;

union V16U { v16h v; v8h h[2]; };

#define SBS 20  // score buffer column stride (floats): 16 rows + 4 pad (16B aligned)

// Fragment-major address for 16-bit B operands:
// per 16-column tile, lane L reads 16 contiguous halves at ((k>>4)*16 + (L&15))*16.
// ntStride = Kdepth*16 halves per 16-n tile.
__device__ __forceinline__ int bfrag(int k, int n, int ntStride) {
  return (n >> 4) * ntStride + ((((k >> 4) << 4) + (n & 15)) << 4) + (k & 15);
}

// ---------------- weight f32 -> f16 ----------------
__global__ __launch_bounds__(256) void cvt_f32_f16(const float* __restrict__ s,
                                                   _Float16* __restrict__ d, int n) {
  int i = blockIdx.x * 256 + threadIdx.x;
  if (i < n) d[i] = (_Float16)s[i];
}

// ---------------- embed: x(b,6,s,t) -> X32(b,s,t,512) = x@proj_w + proj_b + pos ----
__global__ __launch_bounds__(128) void embed_kernel(const float* __restrict__ xin,
                                                    const float* __restrict__ pos,
                                                    const float* __restrict__ pw,
                                                    const float* __restrict__ pbv,
                                                    float* __restrict__ X32) {
  int row = blockIdx.x;
  int bs = row / TP;
  int t = row - bs * TP;
  int b = bs / Ss, s = bs - b * Ss;
  bool valid = (t < Tt);
  float xi[6];
  if (valid) {
#pragma unroll
    for (int i = 0; i < 6; ++i)
      xi[i] = xin[(((size_t)b * 6 + i) * Ss + s) * Tt + t];
  }
#pragma unroll
  for (int u = 0; u < 4; ++u) {
    int dd = threadIdx.x + 128 * u;
    float v = 0.f;
    if (valid) {
      v = pbv[dd] + pos[(size_t)t * Dm + dd];
#pragma unroll
      for (int i = 0; i < 6; ++i) v += xi[i] * pw[i * Dm + dd];
    }
    X32[(size_t)row * Dm + dd] = v;  // padded rows (t>=300) zeroed
  }
}

// ---------------- layernorm over 512, strided row gather ----------------
template <int F16OUT>
__global__ __launch_bounds__(128) void ln_kernel(const float* __restrict__ src,
                                                 _Float16* __restrict__ d16,
                                                 float* __restrict__ d32,
                                                 const float* __restrict__ g,
                                                 const float* __restrict__ bvec,
                                                 int rowsPerBs, int step, int validT) {
  int blk = blockIdx.x;
  int bs = blk / rowsPerBs;
  int zi = blk - bs * rowsPerBs;
  int t = zi * step;
  bool valid = (t < validT);
  const float* x = src + ((size_t)bs * TP + t) * Dm;
  int lane = threadIdx.x & 31, wave = threadIdx.x >> 5;
  float v[4];
  float sum = 0.f, sq = 0.f;
#pragma unroll
  for (int u = 0; u < 4; ++u) {
    int dd = threadIdx.x + 128 * u;
    v[u] = valid ? x[dd] : 0.f;
    sum += v[u];
    sq += v[u] * v[u];
  }
#pragma unroll
  for (int m = 16; m; m >>= 1) {
    sum += __shfl_xor(sum, m, 32);
    sq += __shfl_xor(sq, m, 32);
  }
  __shared__ float rs[4], rq[4];
  if (lane == 0) { rs[wave] = sum; rq[wave] = sq; }
  __syncthreads();
  if (threadIdx.x == 0) {
    float a = rs[0] + rs[1] + rs[2] + rs[3];
    float a2 = rq[0] + rq[1] + rq[2] + rq[3];
    float mean = a * (1.f / Dm);
    float var = a2 * (1.f / Dm) - mean * mean;
    rs[0] = mean;
    rq[0] = rsqrtf(var + 1e-5f);
  }
  __syncthreads();
  float mean = rs[0], inv = rq[0];
  size_t drow = (size_t)bs * rowsPerBs + zi;
#pragma unroll
  for (int u = 0; u < 4; ++u) {
    int dd = threadIdx.x + 128 * u;
    float y = (v[u] - mean) * inv * g[dd] + bvec[dd];
    if (F16OUT) d16[drow * Dm + dd] = (_Float16)y;
    else        d32[drow * Dm + dd] = y;
  }
}

// ---------------- generic WMMA GEMM: C(MxN) = A(MxK) * B(KxN) + epilogue ----------
// Block tile 64x64, 8 waves (2 M-groups x 4 N-tiles), 2 accumulators/wave.
enum { EPI_F16 = 0, EPI_KV = 1, EPI_P1 = 2, EPI_GELU = 3, EPI_ADD = 4 };

template <int EPI, int AF32>
__global__ __launch_bounds__(256) void gemm_wmma(const void* __restrict__ Ap,
                                                 const _Float16* __restrict__ Bw,
                                                 const float* __restrict__ bias,
                                                 const float* __restrict__ res32,
                                                 _Float16* __restrict__ o16a,
                                                 _Float16* __restrict__ o16b,
                                                 float* __restrict__ o32,
                                                 int K, int N) {
  const int lane = threadIdx.x & 31;
  const int wave = threadIdx.x >> 5;
  const int wm = wave >> 2;   // 0..1
  const int wn = wave & 3;    // 0..3
  const int m0 = blockIdx.x * 64;
  const int n0 = blockIdx.y * 64;
  const int kh = lane >> 4;
  const int l15 = lane & 15;
  __shared__ _Float16 As[64][40];   // 64x32 A tile (+pad), row-major
  __shared__ _Float16 Bs[2048];     // 32x64 B tile, fragment-major (ntStride=512)
  v8f acc0 = {}, acc1 = {};
  const int raA = threadIdx.x >> 2;        // 0..63
  const int caA = (threadIdx.x & 3) * 8;   // 8 halves per thread
  const int raB = threadIdx.x >> 3;        // 0..31
  const int cbB = (threadIdx.x & 7) * 8;
  for (int k0 = 0; k0 < K; k0 += 32) {
    // stage A (row-major)
    v8h av;
    if (AF32) {
      const float* Af = (const float*)Ap;
#pragma unroll
      for (int i = 0; i < 8; ++i)
        av[i] = (_Float16)Af[(size_t)(m0 + raA) * K + k0 + caA + i];
    } else {
      const _Float16* Ah = (const _Float16*)Ap;
      av = *(const v8h*)&Ah[(size_t)(m0 + raA) * K + k0 + caA];
    }
    *(v8h*)&As[raA][caA] = av;
    // stage B (coalesced 128b global read, fragment-major LDS scatter)
    v8h bv = *(const v8h*)&Bw[(size_t)(k0 + raB) * N + n0 + cbB];
#pragma unroll
    for (int i = 0; i < 8; ++i) Bs[bfrag(raB, cbB + i, 512)] = bv[i];
    // prefetch next K-step tiles into near caches (WGP-scope prefetch)
    if (k0 + 32 < K) {
      __builtin_prefetch(&Bw[(size_t)(k0 + 32 + raB) * N + n0 + cbB], 0, 3);
      if (AF32)
        __builtin_prefetch(&((const float*)Ap)[(size_t)(m0 + raA) * K + k0 + 32 + caA], 0, 3);
      else
        __builtin_prefetch(&((const _Float16*)Ap)[(size_t)(m0 + raA) * K + k0 + 32 + caA], 0, 3);
    }
    __syncthreads();
    // fragments: A = two contiguous 16B runs per lane; B = one 32B run per lane
    V16U a0, a1, bf;
    const int am0 = 16 * wm + l15;
    const int am1 = am0 + 32;
    a0.h[0] = *(const v8h*)&As[am0][16 * kh];
    a0.h[1] = *(const v8h*)&As[am0][16 * kh + 8];
    a1.h[0] = *(const v8h*)&As[am1][16 * kh];
    a1.h[1] = *(const v8h*)&As[am1][16 * kh + 8];
    const _Float16* bb = &Bs[wn * 512 + (((kh << 4) + l15) << 4)];
    bf.h[0] = *(const v8h*)bb;
    bf.h[1] = *(const v8h*)(bb + 8);
    acc0 = __builtin_amdgcn_wmma_f32_16x16x32_f16(false, a0.v, false, bf.v, (short)0,
                                                  acc0, false, false);
    acc1 = __builtin_amdgcn_wmma_f32_16x16x32_f16(false, a1.v, false, bf.v, (short)0,
                                                  acc1, false, false);
    __syncthreads();
  }
#pragma unroll
  for (int tt = 0; tt < 2; ++tt) {
    v8f acc = tt ? acc1 : acc0;
    const int crow0 = m0 + 16 * wm + 32 * tt + (kh << 3);
    const int ccol = n0 + 16 * wn + l15;
#pragma unroll
    for (int j = 0; j < 8; ++j) {
      int r = crow0 + j;
      float v = acc[j];
      if (EPI == EPI_F16 || EPI == EPI_KV || EPI == EPI_P1) v += bias[ccol];
      if (EPI == EPI_F16) o16a[(size_t)r * N + ccol] = (_Float16)v;
      if (EPI == EPI_KV) {  // interleaved kv: even col -> K, odd col -> V (ld N/2)
        int c2 = ccol >> 1;
        if (ccol & 1) o16b[(size_t)r * (N >> 1) + c2] = (_Float16)v;
        else          o16a[(size_t)r * (N >> 1) + c2] = (_Float16)v;
      }
      if (EPI == EPI_P1) {
        v += res32[(size_t)r * N + ccol];
        o32[(size_t)r * N + ccol] = v;
      }
      if (EPI == EPI_GELU) {
        float gg = 0.5f * v * (1.f + erff(v * 0.70710678118f));
        o16a[(size_t)r * N + ccol] = (_Float16)gg;
      }
      if (EPI == EPI_ADD) {
        size_t idx = (size_t)r * N + ccol;
        o32[idx] = o32[idx] + v;
      }
    }
  }
}

// ---------------- fused per-(b,s,head) attention ----------------
// LDS: Kt frag-major (f16), V frag-major (f16), scores column-major [TKP][SBS] (f32),
//      P row-major [16][TKP] (f16).  Bytes = 368*TKP.
__global__ __launch_bounds__(256) void attn_kernel(const _Float16* __restrict__ Q16,
                                                   const _Float16* __restrict__ Kb,
                                                   const _Float16* __restrict__ Vb,
                                                   _Float16* __restrict__ CAT,
                                                   int tk, int KT, int TKP,
                                                   int catColBase) {
  const int bs = blockIdx.x;
  const int h = blockIdx.y;
  extern __shared__ char smem[];
  _Float16* kt = (_Float16*)smem;                      // 64*TKP halves, frag-major K^T
  _Float16* vt = kt + (size_t)64 * TKP;                // TKP*64 halves, frag-major V
  float* sb = (float*)(vt + (size_t)TKP * 64);         // [TKP][SBS] column-major scores
  _Float16* pbm = (_Float16*)(sb + (size_t)SBS * TKP); // [16][TKP] row-major probs
  const int tid = threadIdx.x;
  const int lane = tid & 31, wave = tid >> 5, kh = lane >> 4;
  const int l15 = lane & 15;
  const int vStride = TKP * 16;
  const _Float16* Kp = Kb + ((size_t)bs * KT) * Dm + h * 64;
  const _Float16* Vp = Vb + ((size_t)bs * KT) * Dm + h * 64;
  for (int i = tid; i < KT * 8; i += 256) {
    int t = i >> 3, c8 = (i & 7) * 8;
    v8h kv = *(const v8h*)(Kp + (size_t)t * Dm + c8);
    v8h vv = *(const v8h*)(Vp + (size_t)t * Dm + c8);
#pragma unroll
    for (int e = 0; e < 8; ++e) {
      int c = c8 + e;
      kt[bfrag(c, t, 1024)] = kv[e];     // K^T: k=c (depth 64), n=t
      vt[bfrag(t, c, vStride)] = vv[e];  // V:  k=t (depth TKP), n=c
    }
  }
  for (int i = tid; i < (TKP - KT) * 8; i += 256) {  // zero pad keys
    int t = KT + (i >> 3), c8 = (i & 7) * 8;
#pragma unroll
    for (int e = 0; e < 8; ++e) {
      kt[bfrag(c8 + e, t, 1024)] = (_Float16)0.f;
      vt[bfrag(t, c8 + e, vStride)] = (_Float16)0.f;
    }
  }
  __syncthreads();
  const _Float16* Qp = Q16 + ((size_t)bs * TP) * Dm + h * 64;
  const int numNT = KT >> 4;
  for (int qt = 0; qt < TP / 16; ++qt) {
    const int q0 = qt * 16;
    V16U a0, a1;  // Q fragments for k0 = 0 and 32 (contiguous 16B runs)
    {
      const _Float16* qrow = Qp + (size_t)(q0 + l15) * Dm;
      a0.h[0] = *(const v8h*)(qrow + 16 * kh);
      a0.h[1] = *(const v8h*)(qrow + 8 + 16 * kh);
      a1.h[0] = *(const v8h*)(qrow + 32 + 16 * kh);
      a1.h[1] = *(const v8h*)(qrow + 40 + 16 * kh);
      if (qt + 1 < TP / 16)  // prefetch next Q tile into near caches
        __builtin_prefetch(qrow + Dm * 16, 0, 3);
    }
    // scores S = Q(16x64) * K^T(64xKT), tiles distributed across 8 waves
    for (int nt = wave; nt < numNT; nt += 8) {
      v8f c = {};
      V16U b0, b1;
      const _Float16* kb0 = kt + nt * 1024 + (((kh << 4) + l15) << 4);
      const _Float16* kb1 = kt + nt * 1024 + ((((2 + kh) << 4) + l15) << 4);
      b0.h[0] = *(const v8h*)kb0;
      b0.h[1] = *(const v8h*)(kb0 + 8);
      b1.h[0] = *(const v8h*)kb1;
      b1.h[1] = *(const v8h*)(kb1 + 8);
      c = __builtin_amdgcn_wmma_f32_16x16x32_f16(false, a0.v, false, b0.v, (short)0,
                                                 c, false, false);
      c = __builtin_amdgcn_wmma_f32_16x16x32_f16(false, a1.v, false, b1.v, (short)0,
                                                 c, false, false);
      // C-tile writeback: lane's 8 rows are contiguous in column-major sb
      const int cr0 = kh << 3;
      const int cc = nt * 16 + l15;
      v4f lo, hi;
#pragma unroll
      for (int j = 0; j < 4; ++j) {
        lo[j] = c[j] * 0.0625f;      // scale = (512//2)^-0.5 = 1/16
        hi[j] = c[4 + j] * 0.0625f;
      }
      *(v4f*)&sb[cc * SBS + cr0] = lo;
      *(v4f*)&sb[cc * SBS + cr0 + 4] = hi;
    }
    __syncthreads();
    {  // masked softmax over key dim, 16 threads per row
      const int row = tid >> 4, sub = tid & 15;
      float mx = -3.0e38f;
      for (int cc = sub; cc < tk; cc += 16) mx = fmaxf(mx, sb[cc * SBS + row]);
#pragma unroll
      for (int m = 8; m; m >>= 1) mx = fmaxf(mx, __shfl_xor(mx, m, 16));
      float sum = 0.f;
      for (int cc = sub; cc < tk; cc += 16) {
        float e = __expf(sb[cc * SBS + row] - mx);
        sb[cc * SBS + row] = e;
        sum += e;
      }
#pragma unroll
      for (int m = 8; m; m >>= 1) sum += __shfl_xor(sum, m, 16);
      float inv = 1.f / sum;
      for (int cc = sub; cc < tk; cc += 16)
        pbm[row * TKP + cc] = (_Float16)(sb[cc * SBS + row] * inv);
      for (int cc = tk + sub; cc < TKP; cc += 16)
        pbm[row * TKP + cc] = (_Float16)0.f;  // mask pad keys
    }
    __syncthreads();
    // O = P(16xTKP) * V(TKPx64), 4 waves own one 16-col tile each
    if (wave < 4) {
      v8f o = {};
      for (int k0 = 0; k0 < TKP; k0 += 32) {
        V16U ap, bv;
        const _Float16* prow = pbm + l15 * TKP + k0;
        ap.h[0] = *(const v8h*)(prow + 16 * kh);
        ap.h[1] = *(const v8h*)(prow + 8 + 16 * kh);
        const int kt4 = (k0 >> 4) + kh;
        const _Float16* vb = vt + wave * vStride + (((kt4 << 4) + l15) << 4);
        bv.h[0] = *(const v8h*)vb;
        bv.h[1] = *(const v8h*)(vb + 8);
        o = __builtin_amdgcn_wmma_f32_16x16x32_f16(false, ap.v, false, bv.v, (short)0,
                                                   o, false, false);
      }
      const int cr0 = kh << 3;
      const int cc = catColBase + h * 64 + wave * 16 + l15;
      const size_t rb = (size_t)bs * TP + q0;
#pragma unroll
      for (int j = 0; j < 8; ++j)
        CAT[(rb + cr0 + j) * 1536 + cc] = (_Float16)o[j];
    }
    __syncthreads();
  }
}

// ---------------- final proj2 (512 -> 6) with output transpose ----------------
__global__ __launch_bounds__(256) void proj2_kernel(const float* __restrict__ O32,
                                                    const float* __restrict__ w,
                                                    const float* __restrict__ b,
                                                    float* __restrict__ out) {
  int gid = blockIdx.x * 256 + threadIdx.x;
  const int tot = Bb * Ss * Tt * 6;
  if (gid >= tot) return;
  int j = gid % 6;
  int t = (gid / 6) % Tt;
  int bs = gid / (6 * Tt);
  const float* r = O32 + ((size_t)bs * TP + t) * Dm;
  float acc = b[j];
  for (int d = 0; d < Dm; ++d) acc += r[d] * w[d * 6 + j];
  int bb = bs / Ss, ss = bs - bb * Ss;
  out[(((size_t)bb * 6 + j) * Ss + ss) * Tt + t] = acc;
}

// ---------------- host ----------------
extern "C" void kernel_launch(void* const* d_in, const int* in_sizes, int n_in,
                              void* d_out, int out_size, void* d_ws, size_t ws_size,
                              hipStream_t stream) {
  (void)in_sizes; (void)n_in; (void)out_size; (void)ws_size;
  const float* x_in = (const float*)d_in[0];
  const float* pos = (const float*)d_in[1];
  const float* proj_w = (const float*)d_in[2];
  const float* proj_b = (const float*)d_in[3];
  const float* q_w = (const float*)d_in[4];
  const float* q_b = (const float*)d_in[5];
  const float* kva_w = (const float*)d_in[6];
  const float* kva_b = (const float*)d_in[7];
  const float* kvh_w = (const float*)d_in[8];
  const float* kvh_b = (const float*)d_in[9];
  const float* kvf_w = (const float*)d_in[10];
  const float* kvf_b = (const float*)d_in[11];
  const float* lna_g = (const float*)d_in[12];
  const float* lna_b = (const float*)d_in[13];
  const float* lnh_g = (const float*)d_in[14];
  const float* lnh_b = (const float*)d_in[15];
  const float* lnf_g = (const float*)d_in[16];
  const float* lnf_b = (const float*)d_in[17];
  const float* p1_w = (const float*)d_in[18];
  const float* p1_b = (const float*)d_in[19];
  const float* ln2_g = (const float*)d_in[20];
  const float* ln2_b = (const float*)d_in[21];
  const float* fc1_w = (const float*)d_in[22];
  const float* fc2_w = (const float*)d_in[23];
  const float* p2_w = (const float*)d_in[24];
  const float* p2_b = (const float*)d_in[25];

  const int BS = Bb * Ss;                      // 192
  const size_t R = (size_t)BS * TP;            // 58368 padded rows
  const int KTA = 304, KTH = 160, KTF = 80;    // padded key counts per branch
  const size_t RA = (size_t)BS * KTA, RH = (size_t)BS * KTH, RF = (size_t)BS * KTF;

  char* base = (char*)d_ws;
  size_t off = 0;
  auto alloc = [&](size_t bytes) -> char* {
    char* p = base + off;
    off += (bytes + 255) & ~(size_t)255;
    return p;
  };
  float* X32 = (float*)alloc(R * Dm * 4);       // x / residual; later reused as y32
  float* OUT32 = (float*)alloc(R * Dm * 4);
  _Float16* Q16 = (_Float16*)alloc(R * Dm * 2);
  _Float16* ZA = (_Float16*)alloc(RA * Dm * 2); // later reused as gelu output G16
  _Float16* ZH = (_Float16*)alloc(RH * Dm * 2);
  _Float16* ZF = (_Float16*)alloc(RF * Dm * 2);
  _Float16* KA = (_Float16*)alloc(RA * Dm * 2);
  _Float16* VA = (_Float16*)alloc(RA * Dm * 2);
  _Float16* KH = (_Float16*)alloc(RH * Dm * 2);
  _Float16* VH = (_Float16*)alloc(RH * Dm * 2);
  _Float16* KF = (_Float16*)alloc(RF * Dm * 2);
  _Float16* VF = (_Float16*)alloc(RF * Dm * 2);
  _Float16* CAT = (_Float16*)alloc(R * 1536 * 2);
  _Float16* qW16 = (_Float16*)alloc((size_t)Dm * Dm * 2);
  _Float16* kvaW16 = (_Float16*)alloc((size_t)Dm * 2 * Dm * 2);
  _Float16* kvhW16 = (_Float16*)alloc((size_t)Dm * 2 * Dm * 2);
  _Float16* kvfW16 = (_Float16*)alloc((size_t)Dm * 2 * Dm * 2);
  _Float16* p1W16 = (_Float16*)alloc((size_t)3 * Dm * Dm * 2);
  _Float16* fc1W16 = (_Float16*)alloc((size_t)Dm * Dm * 2);
  _Float16* fc2W16 = (_Float16*)alloc((size_t)Dm * Dm * 2);
  _Float16* G16 = ZA;  // reuse (ZA free after kv GEMMs)
  float* Y32 = X32;    // reuse (X32 residual consumed by proj1 epilogue)

  auto cvt = [&](const float* s, _Float16* d, int n) {
    cvt_f32_f16<<<(n + 255) / 256, 256, 0, stream>>>(s, d, n);
  };
  cvt(q_w, qW16, Dm * Dm);
  cvt(kva_w, kvaW16, Dm * 2 * Dm);
  cvt(kvh_w, kvhW16, Dm * 2 * Dm);
  cvt(kvf_w, kvfW16, Dm * 2 * Dm);
  cvt(p1_w, p1W16, 3 * Dm * Dm);
  cvt(fc1_w, fc1W16, Dm * Dm);
  cvt(fc2_w, fc2W16, Dm * Dm);

  embed_kernel<<<(unsigned)R, 128, 0, stream>>>(x_in, pos, proj_w, proj_b, X32);

  ln_kernel<1><<<(unsigned)RA, 128, 0, stream>>>(X32, ZA, nullptr, lna_g, lna_b, KTA, 1, Tt);
  ln_kernel<1><<<(unsigned)RH, 128, 0, stream>>>(X32, ZH, nullptr, lnh_g, lnh_b, KTH, 2, Tt);
  ln_kernel<1><<<(unsigned)RF, 128, 0, stream>>>(X32, ZF, nullptr, lnf_g, lnf_b, KTF, 4, Tt);

  // q = x @ q_w + q_b (A from f32, converted while staging into LDS)
  gemm_wmma<EPI_F16, 1><<<dim3((unsigned)(R / 64), Dm / 64), 256, 0, stream>>>(
      X32, qW16, q_b, nullptr, Q16, nullptr, nullptr, Dm, Dm);
  // kv projections: interleaved 1024-wide output deinterleaved to K/V (ld 512)
  gemm_wmma<EPI_KV, 0><<<dim3((unsigned)(RA / 64), 16), 256, 0, stream>>>(
      ZA, kvaW16, kva_b, nullptr, KA, VA, nullptr, Dm, 2 * Dm);
  gemm_wmma<EPI_KV, 0><<<dim3((unsigned)(RH / 64), 16), 256, 0, stream>>>(
      ZH, kvhW16, kvh_b, nullptr, KH, VH, nullptr, Dm, 2 * Dm);
  gemm_wmma<EPI_KV, 0><<<dim3((unsigned)(RF / 64), 16), 256, 0, stream>>>(
      ZF, kvfW16, kvf_b, nullptr, KF, VF, nullptr, Dm, 2 * Dm);

  // fused attention per (b,s,head): LDS bytes = 368 * TKP
  hipFuncSetAttribute(reinterpret_cast<const void*>(attn_kernel),
                      hipFuncAttributeMaxDynamicSharedMemorySize, 368 * 320);
  attn_kernel<<<dim3(BS, Hh), 256, 368 * 320, stream>>>(Q16, KA, VA, CAT, Tt, KTA, 320, 0);
  attn_kernel<<<dim3(BS, Hh), 256, 368 * 160, stream>>>(Q16, KH, VH, CAT, Tt / 2, KTH, 160, 512);
  attn_kernel<<<dim3(BS, Hh), 256, 368 * 96, stream>>>(Q16, KF, VF, CAT, Tt / 4, KTF, 96, 1024);

  // proj1 + residual
  gemm_wmma<EPI_P1, 0><<<dim3((unsigned)(R / 64), Dm / 64), 256, 0, stream>>>(
      CAT, p1W16, p1_b, X32, nullptr, nullptr, OUT32, 3 * Dm, Dm);
  // LN2 -> Y32 (reuses X32 storage)
  ln_kernel<0><<<(unsigned)R, 128, 0, stream>>>(OUT32, nullptr, Y32, ln2_g, ln2_b, TP, 1, TP);
  // fc1 + exact gelu
  gemm_wmma<EPI_GELU, 1><<<dim3((unsigned)(R / 64), Dm / 64), 256, 0, stream>>>(
      Y32, fc1W16, nullptr, nullptr, G16, nullptr, nullptr, Dm, Dm);
  // fc2, accumulate into OUT32
  gemm_wmma<EPI_ADD, 0><<<dim3((unsigned)(R / 64), Dm / 64), 256, 0, stream>>>(
      G16, fc2W16, nullptr, nullptr, nullptr, nullptr, OUT32, Dm, Dm);

  // proj2 + transpose to (b,6,s,t)
  const int tot = Bb * Ss * Tt * 6;
  proj2_kernel<<<(tot + 255) / 256, 256, 0, stream>>>(OUT32, p2_w, p2_b, (float*)d_out);
}
